// SAGELayer_37323265802325
// MI455X (gfx1250) — matrix operand order repaired and checked by reading, not compiled, not run on hardware
//
#include <hip/hip_runtime.h>

typedef __attribute__((ext_vector_type(16))) __bf16 v16bf;
typedef __attribute__((ext_vector_type(8)))  float  v8f;
typedef __attribute__((ext_vector_type(4)))  float  f32x4;

#define IN_F   256
#define OUT_F  256
#define KDIM   512          // 2*IN_F
#define NSAMP  25
#define ROWS   64           // batch rows per workgroup
#define LDS_STRIDE 520      // 512 + 8 ushort pad -> conflict-free ds_load_b128

union FragBF { uint4 u[2]; v16bf v; };

__device__ __forceinline__ unsigned short f2bf(float x) {
    union { float f; unsigned int u; } v; v.f = x;
    unsigned int r = v.u + 0x7FFFu + ((v.u >> 16) & 1u);   // RNE
    return (unsigned short)(r >> 16);
}
__device__ __forceinline__ unsigned int pack2(float x, float y) {
    return (unsigned)f2bf(x) | ((unsigned)f2bf(y) << 16);
}
__device__ __forceinline__ void store4bf16(unsigned short* p, f32x4 v) {
    uint2 d;
    d.x = pack2(v[0], v[1]);
    d.y = pack2(v[2], v[3]);
    *(uint2*)p = d;
}

// ---------------------------------------------------------------------------
// W [OUT_F][KDIM] f32 -> bf16, pre-swizzled into WMMA B-fragment order.
// 16B group index: g = ((nt*16 + ks)*2 + h)*32 + lane
//   n = nt*16 + (lane&15), k0 = ks*32 + (lane>>4)*8 + h*16
// => every phase-2 B load is 32 lanes x 16B fully contiguous (512B dense).
// ---------------------------------------------------------------------------
__global__ __launch_bounds__(256) void sage_prepW(const float* __restrict__ W,
                                                  unsigned short* __restrict__ Wb) {
    const int g    = blockIdx.x * 256 + threadIdx.x;   // 0..16383
    const int lane = g & 31;
    const int h    = (g >> 5) & 1;
    const int ks   = (g >> 6) & 15;
    const int nt   = g >> 10;                          // 0..15
    const int n    = nt * 16 + (lane & 15);
    const int k0   = ks * 32 + (lane >> 4) * 8 + h * 16;

    const f32x4* src = (const f32x4*)(W + (size_t)n * KDIM + k0);
    f32x4 a = src[0], b = src[1];
    uint4 d;
    d.x = pack2(a[0], a[1]);
    d.y = pack2(a[2], a[3]);
    d.z = pack2(b[0], b[1]);
    d.w = pack2(b[2], b[3]);
    ((uint4*)Wb)[g] = d;
}

// ---------------------------------------------------------------------------
// Fused gather + mean + concat + bf16 WMMA GEMM
// ---------------------------------------------------------------------------
__global__ __launch_bounds__(256) void sage_fused(const float* __restrict__ feat,
                                                  const int* __restrict__ nidx,
                                                  const int* __restrict__ gidx,
                                                  const unsigned short* __restrict__ Wb,
                                                  float* __restrict__ out,
                                                  int batch) {
    __shared__ __align__(16) unsigned short lds[ROWS * LDS_STRIDE];

    const int tid = threadIdx.x;
    const int wg  = blockIdx.x;

    // ---------- Phase 1: gather self + mean(25 neighbors) -> bf16 LDS ------
    {
        const int r   = tid >> 2;          // 0..63 : row within tile
        const int c   = tid & 3;           // 4 threads cover 256 feats
        const int row = wg * ROWS + r;

        f32x4 acc[16];
#pragma unroll
        for (int i = 0; i < 16; ++i) acc[i] = (f32x4)(0.f);

        if (row < batch) {
            const float* srow = feat + (size_t)nidx[row] * IN_F;
#pragma unroll
            for (int i = 0; i < 16; ++i) {
                f32x4 v = ((const f32x4*)srow)[c + 4 * i];         // coalesced
                store4bf16(&lds[r * LDS_STRIDE + (c + 4 * i) * 4], v);
            }

            const int* grow = gidx + (size_t)row * NSAMP;
            int nbr[NSAMP];                 // preload indices: one load clause,
#pragma unroll                              // removes per-iter idx->addr dep
            for (int s = 0; s < NSAMP; ++s) nbr[s] = grow[s];

            for (int s = 0; s < NSAMP; ++s) {
                const f32x4* nrow = (const f32x4*)(feat + (size_t)nbr[s] * IN_F);
#pragma unroll
                for (int i = 0; i < 16; ++i) {
                    acc[i] += nrow[c + 4 * i];        // <4 x float> -> v_pk_add_f32
                }
            }
        } else {
            f32x4 z = (f32x4)(0.f);
#pragma unroll
            for (int i = 0; i < 16; ++i)
                store4bf16(&lds[r * LDS_STRIDE + (c + 4 * i) * 4], z);
        }

        const float inv = 1.0f / (float)NSAMP;
#pragma unroll
        for (int i = 0; i < 16; ++i) {
            f32x4 m = acc[i] * inv;
            store4bf16(&lds[r * LDS_STRIDE + IN_F + (c + 4 * i) * 4], m);
        }
    }
    __syncthreads();

    // ---------- Phase 2: WMMA  combined[64x512](LDS) @ W^T(fragment-major) --
    const int wave  = tid >> 5;
    const int lane  = tid & 31;
    const int mtile = wave >> 1;           // 4 M-tiles of 16 rows
    const int ngrp  = wave & 1;            // 2 N-groups of 128 cols
    const int mlane = lane & 15;
    const int kg    = lane >> 4;

    v8f cacc[8] = {};                      // 8 N-tiles of 16x16 f32

    const unsigned short* arow = &lds[(mtile * 16 + mlane) * LDS_STRIDE];
    const uint4* wfrag = (const uint4*)Wb;

    for (int ks = 0; ks < 16; ++ks) {
        const int kb = ks * 32;
        FragBF a;
        // ISA 16-bit A layout: VGPR0-3 = K kb+8*kg+{0..7}, VGPR4-7 = +16
        a.u[0] = *(const uint4*)&arow[kb + kg * 8];
        a.u[1] = *(const uint4*)&arow[kb + 16 + kg * 8];
#pragma unroll
        for (int t = 0; t < 8; ++t) {
            const int nt = ngrp * 8 + t;
            const uint4* bp = wfrag + ((size_t)(nt * 16 + ks) * 2) * 32 + lane;
            FragBF b;
            b.u[0] = bp[0];                // 512B dense across the wave
            b.u[1] = bp[32];
            cacc[t] = __builtin_amdgcn_wmma_f32_16x16x32_bf16(
                false, a.v, false, b.v, (short)0, cacc[t], false, false);
        }
    }

    // ---------- Store D: VGPR j -> m = mtile*16 + j + 8*kg, n = nt*16+mlane -
    const int gm = wg * ROWS + mtile * 16 + 8 * kg;
    float* obase = out + (size_t)gm * OUT_F + ngrp * 128 + mlane;

    if (wg * ROWS + ROWS <= batch) {       // block-uniform fast path
#pragma unroll
        for (int t = 0; t < 8; ++t) {
#pragma unroll
            for (int j = 0; j < 8; ++j) {
                __builtin_nontemporal_store(cacc[t][j],
                                            obase + (size_t)j * OUT_F + t * 16);
            }
        }
    } else {                               // single partial block
#pragma unroll
        for (int t = 0; t < 8; ++t) {
#pragma unroll
            for (int j = 0; j < 8; ++j) {
                if (gm + j < batch)
                    obase[(size_t)j * OUT_F + t * 16] = cacc[t][j];
            }
        }
    }
}

extern "C" void kernel_launch(void* const* d_in, const int* in_sizes, int n_in,
                              void* d_out, int out_size, void* d_ws, size_t ws_size,
                              hipStream_t stream) {
    const float* feat = (const float*)d_in[0];          // [100000,256] f32
    const int*   nidx = (const int*)d_in[1];            // [B] i32
    const int*   gidx = (const int*)d_in[2];            // [B,25] i32
    const float* W    = (const float*)d_in[3];          // [256,512] f32
    float*       out  = (float*)d_out;                  // [B,256] f32
    unsigned short* Wb = (unsigned short*)d_ws;         // swizzled bf16 W (256KB)

    const int batch = in_sizes[1];

    sage_prepW<<<64, 256, 0, stream>>>(W, Wb);          // 16384 fragment groups

    const int nBlocks = (batch + ROWS - 1) / ROWS;
    sage_fused<<<nBlocks, 256, 0, stream>>>(feat, nidx, gidx, Wb, out, batch);
}
